// BiMambaBlock_31628139168142
// MI455X (gfx1250) — matrix operand order repaired and checked by reading, not compiled, not run on hardware
//
#include <hip/hip_runtime.h>
#include <hip/hip_bf16.h>
#include <stdint.h>
#include <stddef.h>

// ---------------- problem constants ----------------
#define BATCH  4
#define SEQ    2048
#define DMODEL 768
#define DINNER 1536          // EXPAND * DMODEL
#define DSTATE 16
#define DTRANK 48
#define MTOK   (BATCH*SEQ)   // 8192 tokens
#define PROJW  (DTRANK + 2*DSTATE)  // 80
#define NCHUNK 16            // scan chunks (parallel scan)
#define CLEN   (SEQ/NCHUNK)  // 128 timesteps per chunk
#define DCH    (DINNER/256)  // 6 channel-chunks per block row

typedef __attribute__((ext_vector_type(8)))  float  v8f;
typedef __attribute__((ext_vector_type(8)))  __bf16 v8bf;
typedef __attribute__((ext_vector_type(16))) __bf16 v16bf;

// ---------------- gfx1250 async global->LDS copy (probe-verified) ----------
#if defined(__HIP_DEVICE_COMPILE__) && \
    __has_builtin(__builtin_amdgcn_global_load_async_to_lds_b128)
#define USE_ASYNC_B 1
typedef __attribute__((__vector_size__(16))) int v4i_t;
typedef __attribute__((address_space(1))) v4i_t as1_v4i;
typedef __attribute__((address_space(3))) v4i_t as3_v4i;
__device__ __forceinline__ void async_ld_b128(const void* g, const void* l) {
  __builtin_amdgcn_global_load_async_to_lds_b128(
      (as1_v4i*)(uintptr_t)g,
      (as3_v4i*)(uint32_t)(uintptr_t)l, 0, 0);
}
__device__ __forceinline__ void wait_async0() {
#if __has_builtin(__builtin_amdgcn_s_wait_asynccnt)
  __builtin_amdgcn_s_wait_asynccnt(0);
#else
  asm volatile("s_wait_asynccnt 0" ::: "memory");
#endif
}
#else
#define USE_ASYNC_B 0
__device__ __forceinline__ void wait_async0() {}
#endif

__device__ __forceinline__ v16bf cat8(v8bf lo, v8bf hi) {
  return __builtin_shufflevector(lo, hi, 0,1,2,3,4,5,6,7,8,9,10,11,12,13,14,15);
}
__device__ __forceinline__ v8bf pack8(float4 a, float4 b) {
  v8bf r;
  r[0] = (__bf16)a.x; r[1] = (__bf16)a.y; r[2] = (__bf16)a.z; r[3] = (__bf16)a.w;
  r[4] = (__bf16)b.x; r[5] = (__bf16)b.y; r[6] = (__bf16)b.z; r[7] = (__bf16)b.w;
  return r;
}
__device__ __forceinline__ v8bf zero8() {
  v8bf r;
#pragma unroll
  for (int i = 0; i < 8; i++) r[i] = (__bf16)0.f;
  return r;
}

// ---------------------------------------------------------------------------
__global__ void cvt_f32_to_bf16(const float* __restrict__ s, __bf16* __restrict__ d, int n) {
  int i = blockIdx.x * 256 + threadIdx.x;
  if (i < n) d[i] = (__bf16)s[i];
}

// ---------------------------------------------------------------------------
// Tiled WMMA GEMM:  C[M,NN] (+)= A[M,KK] * B[NN,KK]^T   (shapes compile-time)
//   128x128x32 tile, 8 waves, each wave 2(M)x4(N) v_wmma_f32_16x16x32_bf16.
//   Double-buffered LDS; A prefetched to regs, B async-copied to LDS.
// ---------------------------------------------------------------------------
template<int EPI, int LDA, int NN, int KK>
__global__ __launch_bounds__(256)
void gemm_bf16_tn(const float* __restrict__ A, int M,
                  const __bf16* __restrict__ Bw,
                  float* __restrict__ C,
                  const float* __restrict__ bias,
                  int flipA, int flipStore, int accumulate)
{
  constexpr bool NGUARD = (NN % 128) != 0;
  constexpr int  KSTEPS = (KK + 31) / 32;

  __shared__ __align__(16) __bf16 As[2][128][40];
  __shared__ __align__(16) __bf16 Bs[2][128][40];

  const int tid = threadIdx.x;
  const int m0  = blockIdx.y * 128;
  const int n0  = blockIdx.x * 128;

  const int lr  = tid >> 1;
  const int seg = (tid & 1) << 4;

  int am = m0 + lr;
  if (flipA) { int t = am % SEQ; am += (SEQ - 1 - t) - t; }
  const float*  arow = A  + (size_t)am * LDA;
  const __bf16* brow = Bw + (size_t)(n0 + lr) * KK;
  const bool nok = NGUARD ? ((n0 + lr) < NN) : true;

  const int wave = tid >> 5;
  const int lane = tid & 31;
  const int hi   = lane >> 4;
  const int ln   = lane & 15;
  const int wm   = (wave & 3) << 5;
  const int wn   = (wave >> 2) << 6;

  v8f acc[2][4] = {};

  float4 aR0, aR1, aR2, aR3;
  v8bf   bR0 = zero8(), bR1 = zero8();
  int    bViaRegs = 0;

  auto loadA_full = [&](int kk) {
    const float4* p = (const float4*)(arow + kk + seg);
    aR0 = p[0]; aR1 = p[1]; aR2 = p[2]; aR3 = p[3];
  };
  auto loadA_tail = [&](int kk) {
    float t[16];
#pragma unroll
    for (int j = 0; j < 16; j++) {
      int k = kk + seg + j;
      t[j] = (k < KK) ? arow[k] : 0.f;
    }
    aR0 = make_float4(t[0], t[1], t[2], t[3]);
    aR1 = make_float4(t[4], t[5], t[6], t[7]);
    aR2 = make_float4(t[8], t[9], t[10], t[11]);
    aR3 = make_float4(t[12], t[13], t[14], t[15]);
  };
  auto storeA = [&](int st) {
    *(v8bf*)&As[st][lr][seg + 0] = pack8(aR0, aR1);
    *(v8bf*)&As[st][lr][seg + 8] = pack8(aR2, aR3);
  };
  auto loadB_full = [&](int st, int kk) {
    const __bf16* p = brow + kk + seg;
#if USE_ASYNC_B
    if (nok) {
      async_ld_b128(p,     &Bs[st][lr][seg]);
      async_ld_b128(p + 8, &Bs[st][lr][seg + 8]);
      bViaRegs = 0;
    } else {
      bR0 = zero8(); bR1 = zero8(); bViaRegs = 1;
    }
#else
    if (nok) { bR0 = ((const v8bf*)p)[0]; bR1 = ((const v8bf*)p)[1]; }
    else     { bR0 = zero8();             bR1 = zero8(); }
    bViaRegs = 1;
#endif
  };
  auto loadB_tail = [&](int kk) {
    v8bf r0 = zero8(), r1 = zero8();
    if (nok) {
#pragma unroll
      for (int q = 0; q < 8; q++) {
        int k0 = kk + seg + q, k1 = kk + seg + 8 + q;
        if (k0 < KK) r0[q] = brow[k0];
        if (k1 < KK) r1[q] = brow[k1];
      }
    }
    bR0 = r0; bR1 = r1; bViaRegs = 1;
  };
  auto storeB = [&](int st) {
    if (bViaRegs) {
      *(v8bf*)&Bs[st][lr][seg + 0] = bR0;
      *(v8bf*)&Bs[st][lr][seg + 8] = bR1;
    }
  };

  loadA_full(0);
  loadB_full(0, 0);
  storeA(0);
  storeB(0);
  wait_async0();
  __syncthreads();

  for (int s = 0; s < KSTEPS; ++s) {
    const int st = s & 1;
    const bool haveNext = (s + 1 < KSTEPS);

    if (haveNext) {
      int kk2 = (s + 1) * 32;
      if constexpr (KK % 32 != 0) {
        if (kk2 + 32 <= KK) { loadA_full(kk2); loadB_full(st ^ 1, kk2); }
        else                { loadA_tail(kk2); loadB_tail(kk2); }
      } else {
        loadA_full(kk2); loadB_full(st ^ 1, kk2);
      }
    }

    v16bf af[2];
#pragma unroll
    for (int i = 0; i < 2; i++) {
      const __bf16* ap = &As[st][wm + (i << 4) + ln][0];
      af[i] = cat8(*(const v8bf*)(ap + 8 * hi), *(const v8bf*)(ap + 16 + 8 * hi));
    }
    v16bf bfr[4];
#pragma unroll
    for (int j = 0; j < 4; j++) {
      const __bf16* bp = &Bs[st][wn + (j << 4) + ln][0];
      bfr[j] = cat8(*(const v8bf*)(bp + 8 * hi), *(const v8bf*)(bp + 16 + 8 * hi));
    }
#pragma unroll
    for (int i = 0; i < 2; i++)
#pragma unroll
      for (int j = 0; j < 4; j++)
        acc[i][j] = __builtin_amdgcn_wmma_f32_16x16x32_bf16(
            false, af[i], false, bfr[j], (short)0, acc[i][j], false, false);

    if (haveNext) {
      storeA(st ^ 1);
      storeB(st ^ 1);
      wait_async0();
      __syncthreads();
    }
  }

#pragma unroll
  for (int i = 0; i < 2; i++) {
#pragma unroll
    for (int j = 0; j < 4; j++) {
      int ncol = n0 + wn + (j << 4) + ln;
      if constexpr (NGUARD) { if (ncol >= NN) continue; }
      float bv = (EPI == 1) ? bias[ncol] : 0.f;
#pragma unroll
      for (int r = 0; r < 8; r++) {
        int m = m0 + wm + (i << 4) + (hi << 3) + r;
        float v = acc[i][j][r];
        if (EPI == 1) { v += bv; v = (v > 20.f) ? v : log1pf(__expf(v)); }
        int sm = m;
        if (flipStore) { int t = m % SEQ; sm += (SEQ - 1 - t) - t; }
        float* cp = C + (size_t)sm * NN + ncol;
        if (accumulate) *cp += v; else *cp = v;
      }
    }
  }
}

// ---------------------------------------------------------------------------
// depthwise causal conv (K=4) + SiLU, 4 channels per thread (float4 I/O)
// ---------------------------------------------------------------------------
__global__ __launch_bounds__(256)
void conv_silu_kernel(const float* __restrict__ xz,
                      const float* __restrict__ cw,
                      const float* __restrict__ cb,
                      float* __restrict__ xconv)
{
  const int QD = DINNER / 4;
  size_t q = (size_t)blockIdx.x * 256 + threadIdx.x;
  int    d = (int)(q % QD) * 4;
  size_t m = q / QD;
  int    t = (int)(m % SEQ);
  const float* base = xz + (m - (size_t)t) * (2 * DINNER) + d;
  float4 acc = *(const float4*)(cb + d);
#pragma unroll
  for (int k = 0; k < 4; k++) {
    int tt = t - 3 + k;
    if (tt >= 0) {
      float4 v = *(const float4*)(base + (size_t)tt * (2 * DINNER));
      acc.x = fmaf(cw[(d + 0) * 4 + k], v.x, acc.x);
      acc.y = fmaf(cw[(d + 1) * 4 + k], v.y, acc.y);
      acc.z = fmaf(cw[(d + 2) * 4 + k], v.z, acc.z);
      acc.w = fmaf(cw[(d + 3) * 4 + k], v.w, acc.w);
    }
  }
  float4 r;
  r.x = acc.x / (1.f + __expf(-acc.x));
  r.y = acc.y / (1.f + __expf(-acc.y));
  r.z = acc.z / (1.f + __expf(-acc.z));
  r.w = acc.w / (1.f + __expf(-acc.w));
  *(float4*)(xconv + m * DINNER + d) = r;
}

// ---------------------------------------------------------------------------
// chunked parallel selective scan (3 passes, linear-recurrence composition):
//   h_t = dA_t o h_{t-1} + (dt*u)_t B_t  =>  over a chunk:
//   h_end = P o h_start + S,  P = prod(dA), S = chunk scan from 0.
// Buffers laid out [b][c][n][d] for coalesced per-n access.
// ---------------------------------------------------------------------------
__global__ __launch_bounds__(256)
void scan_pass1(const float* __restrict__ xconv,
                const float* __restrict__ dtb,
                const float* __restrict__ proj,
                const float* __restrict__ A_log,
                float* __restrict__ Pbuf, float* __restrict__ Sbuf)
{
  int blk = blockIdx.x;
  int dc  = blk % DCH;
  int c   = (blk / DCH) % NCHUNK;
  int b   = blk / (DCH * NCHUNK);
  int d   = dc * 256 + threadIdx.x;

  float A[DSTATE];
#pragma unroll
  for (int n = 0; n < DSTATE; n++) A[n] = -__expf(A_log[d * DSTATE + n]);
  float h[DSTATE], P[DSTATE];
#pragma unroll
  for (int n = 0; n < DSTATE; n++) { h[n] = 0.f; P[n] = 1.f; }

  size_t mbase = (size_t)b * SEQ + (size_t)c * CLEN;
  for (int t = 0; t < CLEN; t++) {
    size_t m = mbase + t;
    const float* bc = proj + m * PROJW + DTRANK;   // B[0..15]
    float dtv = dtb[m * DINNER + d];
    float du  = dtv * xconv[m * DINNER + d];
#pragma unroll
    for (int n = 0; n < DSTATE; n++) {
      float dA = __expf(dtv * A[n]);
      h[n] = fmaf(dA, h[n], du * bc[n]);
      P[n] *= dA;
    }
  }
  size_t o = ((size_t)(b * NCHUNK + c) * DSTATE) * DINNER + d;
#pragma unroll
  for (int n = 0; n < DSTATE; n++) {
    Pbuf[o + (size_t)n * DINNER] = P[n];
    Sbuf[o + (size_t)n * DINNER] = h[n];
  }
}

__global__ __launch_bounds__(256)
void scan_pass2(const float* __restrict__ Pbuf, const float* __restrict__ Sbuf,
                float* __restrict__ Hs)
{
  int b = blockIdx.x / DCH;
  int d = (blockIdx.x % DCH) * 256 + threadIdx.x;
  float hs[DSTATE];
#pragma unroll
  for (int n = 0; n < DSTATE; n++) hs[n] = 0.f;
  for (int c = 0; c < NCHUNK; c++) {
    size_t o = ((size_t)(b * NCHUNK + c) * DSTATE) * DINNER + d;
#pragma unroll
    for (int n = 0; n < DSTATE; n++) {
      size_t idx = o + (size_t)n * DINNER;
      Hs[idx] = hs[n];
      hs[n] = fmaf(Pbuf[idx], hs[n], Sbuf[idx]);
    }
  }
}

__global__ __launch_bounds__(256)
void scan_pass3(const float* __restrict__ xz,
                const float* __restrict__ xconv,
                const float* __restrict__ dtb,
                const float* __restrict__ proj,
                const float* __restrict__ A_log,
                const float* __restrict__ Dv,
                const float* __restrict__ Hs,
                float* __restrict__ ybuf)
{
  int blk = blockIdx.x;
  int dc  = blk % DCH;
  int c   = (blk / DCH) % NCHUNK;
  int b   = blk / (DCH * NCHUNK);
  int d   = dc * 256 + threadIdx.x;

  float A[DSTATE];
#pragma unroll
  for (int n = 0; n < DSTATE; n++) A[n] = -__expf(A_log[d * DSTATE + n]);
  float Dd = Dv[d];
  float h[DSTATE];
  size_t o = ((size_t)(b * NCHUNK + c) * DSTATE) * DINNER + d;
#pragma unroll
  for (int n = 0; n < DSTATE; n++) h[n] = Hs[o + (size_t)n * DINNER];

  size_t mbase = (size_t)b * SEQ + (size_t)c * CLEN;
  for (int t = 0; t < CLEN; t++) {
    size_t m = mbase + t;
    const float* bc = proj + m * PROJW + DTRANK;   // B[0..15], C[0..15]
    float dtv = dtb[m * DINNER + d];
    float u   = xconv[m * DINNER + d];
    float du  = dtv * u;
    float y   = 0.f;
#pragma unroll
    for (int n = 0; n < DSTATE; n++) {
      float dA = __expf(dtv * A[n]);
      h[n] = fmaf(dA, h[n], du * bc[n]);
      y    = fmaf(h[n], bc[DSTATE + n], y);
    }
    float zv = xz[m * (2 * DINNER) + DINNER + d];
    float g  = zv / (1.f + __expf(-zv));
    ybuf[m * DINNER + d] = (y + u * Dd) * g;
  }
}

// ---------------------------------------------------------------------------
// residual add + LayerNorm over DMODEL=768 (one block per token)
// ---------------------------------------------------------------------------
__global__ __launch_bounds__(256)
void ln_kernel(const float* __restrict__ acc, const float* __restrict__ x,
               const float* __restrict__ g, const float* __restrict__ bta,
               float* __restrict__ out)
{
  int m = blockIdx.x;
  int tid = threadIdx.x;
  const float* ar = acc + (size_t)m * DMODEL;
  const float* xr = x   + (size_t)m * DMODEL;
  float v[3]; float s = 0.f, s2 = 0.f;
#pragma unroll
  for (int i = 0; i < 3; i++) {
    int c = tid + i * 256;
    float hv = ar[c] + xr[c];
    v[i] = hv; s += hv; s2 += hv * hv;
  }
#pragma unroll
  for (int off = 16; off > 0; off >>= 1) {
    s  += __shfl_down(s,  off, 32);                // wave32
    s2 += __shfl_down(s2, off, 32);
  }
  __shared__ float red[2][8];
  int wave = tid >> 5, lane = tid & 31;
  if (lane == 0) { red[0][wave] = s; red[1][wave] = s2; }
  __syncthreads();
  s = 0.f; s2 = 0.f;
#pragma unroll
  for (int w = 0; w < 8; w++) { s += red[0][w]; s2 += red[1][w]; }
  float mu  = s  * (1.f / DMODEL);
  float var = s2 * (1.f / DMODEL) - mu * mu;
  float inv = rsqrtf(var + 1e-5f);
  float* orow = out + (size_t)m * DMODEL;
#pragma unroll
  for (int i = 0; i < 3; i++) {
    int c = tid + i * 256;
    orow[c] = (v[i] - mu) * inv * g[c] + bta[c];
  }
}

// ---------------------------------------------------------------------------
extern "C" void kernel_launch(void* const* d_in, const int* in_sizes, int n_in,
                              void* d_out, int out_size, void* d_ws, size_t ws_size,
                              hipStream_t stream)
{
  (void)in_sizes; (void)n_in; (void)out_size; (void)ws_size;
  const float* x    = (const float*)d_in[0];
  const float* ln_g = (const float*)d_in[19];
  const float* ln_b = (const float*)d_in[20];
  float* out = (float*)d_out;

  char* wp = (char*)d_ws;
  auto take = [&](size_t bytes) {
    char* p = wp; wp += (bytes + 255) & ~(size_t)255; return (void*)p;
  };

  float*  xz    = (float*)take((size_t)MTOK * 2 * DINNER * 4);   // 100 MB
  float*  xconv = (float*)take((size_t)MTOK * DINNER * 4);       // 50 MB
  float*  proj  = (float*)take((size_t)MTOK * PROJW * 4);        // 2.6 MB
  float*  dtb   = (float*)take((size_t)MTOK * DINNER * 4);       // 50 MB
  float*  ybuf  = (float*)take((size_t)MTOK * DINNER * 4);       // 50 MB
  float*  accb  = (float*)take((size_t)MTOK * DMODEL * 4);       // 25 MB
  const size_t scanSt = (size_t)BATCH * NCHUNK * DSTATE * DINNER;
  float*  Pbuf  = (float*)take(scanSt * 4);                      // 6.3 MB
  float*  Sbuf  = (float*)take(scanSt * 4);                      // 6.3 MB
  float*  Hsbuf = (float*)take(scanSt * 4);                      // 6.3 MB
  __bf16* Winb  = (__bf16*)take((size_t)2 * DINNER * DMODEL * 2);
  __bf16* Wxb   = (__bf16*)take((size_t)PROJW * DINNER * 2);
  __bf16* Wdtb  = (__bf16*)take((size_t)DINNER * DTRANK * 2);
  __bf16* Woutb = (__bf16*)take((size_t)DMODEL * DINNER * 2);

  for (int dir = 0; dir < 2; dir++) {
    const int base = 1 + dir * 9;
    const float* W_in  = (const float*)d_in[base + 0];
    const float* cw    = (const float*)d_in[base + 1];
    const float* cb    = (const float*)d_in[base + 2];
    const float* W_x   = (const float*)d_in[base + 3];
    const float* W_dt  = (const float*)d_in[base + 4];
    const float* b_dt  = (const float*)d_in[base + 5];
    const float* A_log = (const float*)d_in[base + 6];
    const float* Dv    = (const float*)d_in[base + 7];
    const float* W_out = (const float*)d_in[base + 8];

    const int nWin = 2 * DINNER * DMODEL, nWx = PROJW * DINNER,
              nWdt = DINNER * DTRANK,     nWout = DMODEL * DINNER;
    cvt_f32_to_bf16<<<(nWin  + 255) / 256, 256, 0, stream>>>(W_in,  Winb,  nWin);
    cvt_f32_to_bf16<<<(nWx   + 255) / 256, 256, 0, stream>>>(W_x,   Wxb,   nWx);
    cvt_f32_to_bf16<<<(nWdt  + 255) / 256, 256, 0, stream>>>(W_dt,  Wdtb,  nWdt);
    cvt_f32_to_bf16<<<(nWout + 255) / 256, 256, 0, stream>>>(W_out, Woutb, nWout);

    // xz = (flip? x) @ W_in^T               M=8192 N=3072 K=768
    gemm_bf16_tn<0, DMODEL, 2 * DINNER, DMODEL>
        <<<dim3((2 * DINNER) / 128, MTOK / 128), 256, 0, stream>>>(
        x, MTOK, Winb, xz, nullptr, dir, 0, 0);
    // causal conv + SiLU
    conv_silu_kernel<<<(MTOK * (DINNER / 4)) / 256, 256, 0, stream>>>(xz, cw, cb, xconv);
    // proj = xconv @ W_x^T                  M=8192 N=80 K=1536
    gemm_bf16_tn<0, DINNER, PROJW, DINNER>
        <<<dim3(1, MTOK / 128), 256, 0, stream>>>(
        xconv, MTOK, Wxb, proj, nullptr, 0, 0, 0);
    // dt = softplus(dtr @ W_dt^T + b_dt)    M=8192 N=1536 K=48
    gemm_bf16_tn<1, PROJW, DINNER, DTRANK>
        <<<dim3(DINNER / 128, MTOK / 128), 256, 0, stream>>>(
        proj, MTOK, Wdtb, dtb, b_dt, 0, 0, 0);
    // chunked parallel selective scan
    scan_pass1<<<BATCH * NCHUNK * DCH, 256, 0, stream>>>(
        xconv, dtb, proj, A_log, Pbuf, Sbuf);
    scan_pass2<<<BATCH * DCH, 256, 0, stream>>>(Pbuf, Sbuf, Hsbuf);
    scan_pass3<<<BATCH * NCHUNK * DCH, 256, 0, stream>>>(
        xz, xconv, dtb, proj, A_log, Dv, Hsbuf, ybuf);
    // acc (+)= y @ W_out^T (bwd: flipped rows, accumulate)   M=8192 N=768 K=1536
    gemm_bf16_tn<0, DINNER, DMODEL, DINNER>
        <<<dim3(DMODEL / 128, MTOK / 128), 256, 0, stream>>>(
        ybuf, MTOK, Woutb, accb, nullptr, 0, dir, dir);
  }

  // out = LayerNorm(acc + x)
  ln_kernel<<<MTOK, 256, 0, stream>>>(accb, x, ln_g, ln_b, out);
}
